// MGUNet_25357486916077
// MI455X (gfx1250) — matrix operand order repaired and checked by reading, not compiled, not run on hardware
//
#include <hip/hip_runtime.h>

#define B_    128
#define S_    512
#define E_    512
#define H_    1024
#define NCLS_ 1000

typedef __attribute__((ext_vector_type(16))) __bf16 v16bf;
typedef __attribute__((ext_vector_type(8)))  float  v8f;
typedef __attribute__((ext_vector_type(4)))  unsigned v4u;

union FragBF {
  v16bf f;
  unsigned u[8];
  v4u q[2];
};

// ---- bf16 <-> f32 as raw bits (round-to-nearest-even) ----
__device__ __forceinline__ unsigned short f2bf(float x) {
  unsigned u = __float_as_uint(x);
  u += 0x7FFFu + ((u >> 16) & 1u);
  return (unsigned short)(u >> 16);
}
__device__ __forceinline__ float bf2f(unsigned short h) {
  return __uint_as_float(((unsigned)h) << 16);
}

// LDS byte-offset of a generic pointer that points into LDS
__device__ __forceinline__ unsigned lds_offset_of(const void* p) {
  return (unsigned)(unsigned long long)(__attribute__((address_space(3))) const char*)p;
}

// ---- A fragment (16x32 bf16) from LDS tile (dword-packed, padded pitch) ----
// ISA 7.12.2: lanes 0-15 rows M=0..15 hold K={0..7,16..23}, lanes 16-31 same
// rows hold K={8..15,24..31}; VGPR v<4 -> pair (2v,2v+1)+khalf, v>=4 -> +16.
__device__ __forceinline__ v16bf frag_A_lds(const unsigned* sm, int pitch_dw,
                                            int k0, int lane) {
  FragBF a;
  const int row = lane & 15;
  const int kh2 = (lane < 16) ? 0 : 4;              // khalf(8 halves)/2 dwords
  const unsigned* base = sm + row * pitch_dw + (k0 >> 1);
#pragma unroll
  for (int vi = 0; vi < 4; ++vi) {
    a.u[vi]     = base[kh2 + vi];
    a.u[4 + vi] = base[8 + kh2 + vi];
  }
  return a.f;
}

// ---- B fragment (32x16 bf16) from column-major (transposed) weights ----
// lanes 0-15: col=lane, K={0..7,16..23} ; lanes 16-31: col=lane-16, K={8..15,24..31}.
__device__ __forceinline__ v16bf frag_B_t(const unsigned short* Bt, int ldb,
                                          int k0, int n0, int lane) {
  FragBF b;
  const int col   = n0 + (lane & 15);
  const int khalf = (lane < 16) ? 0 : 8;
  const v4u* base = (const v4u*)(Bt + (size_t)col * ldb + k0 + khalf);
  b.q[0] = base[0];
  b.q[1] = base[2];
  return b.f;
}

#define WMMA_BF16(A, Bf, C) \
  __builtin_amdgcn_wmma_f32_16x16x32_bf16(false, (A), false, (Bf), (short)0, (C), false, false)

// =====================================================================
// Weight convert + transpose into bf16 workspace
// =====================================================================
__global__ void __launch_bounds__(256) mgu_prep_w(
    const float* __restrict__ Wz, const float* __restrict__ Wh,
    const float* __restrict__ Wfc,
    unsigned short* __restrict__ Wx_t, unsigned short* __restrict__ Wr_t,
    unsigned short* __restrict__ Wfc_t) {
  const int NX = 2 * H_ * E_;
  const int NR = 2 * H_ * H_;
  const int NF = H_ * H_;
  int idx = blockIdx.x * 256 + threadIdx.x;
  if (idx < NX) {
    int nn = idx / E_, k = idx % E_;
    float v = (nn < H_) ? Wz[(size_t)k * H_ + nn]
                        : Wh[(size_t)k * H_ + (nn - H_)];
    Wx_t[idx] = f2bf(v);
  } else if (idx < NX + NR) {
    int j = idx - NX;
    int nn = j / H_, k = j % H_;
    float v = (nn < H_) ? Wz[(size_t)(E_ + k) * H_ + nn]
                        : Wh[(size_t)(E_ + k) * H_ + (nn - H_)];
    Wr_t[j] = f2bf(v);
  } else if (idx < NX + NR + NF) {
    int j = idx - NX - NR;
    int nn = j / H_, k = j % H_;
    float v = (nn < NCLS_) ? Wfc[(size_t)k * NCLS_ + nn] : 0.0f;
    Wfc_t[j] = f2bf(v);
  }
}

// =====================================================================
// Embedding gather -> bf16, reordered to [S][B][E] so GEMM rows = s*B+b
// =====================================================================
__global__ void __launch_bounds__(256) mgu_embed(
    const int* __restrict__ x, const float* __restrict__ emb,
    unsigned short* __restrict__ e_bf) {
  const int row = blockIdx.x;          // s*B + b
  const int s = row >> 7;
  const int b = row & (B_ - 1);
  const int tok = x[b * S_ + s];
  const float* src = emb + (size_t)tok * E_;
  unsigned short* dst = e_bf + (size_t)row * E_;
  for (int i = threadIdx.x; i < E_; i += 256) dst[i] = f2bf(src[i]);
}

__global__ void __launch_bounds__(256) mgu_init_h(
    float* __restrict__ h32, unsigned short* __restrict__ hbf) {
  int i = blockIdx.x * 256 + threadIdx.x;
  if (i < B_ * H_) { h32[i] = 0.0f; hbf[i] = 0; }
}

// =====================================================================
// Phase 1: xz/xh = e @ [Wzx | Whx] + bias   (M=S*B=65536, K=E, N=2H)
// =====================================================================
__global__ void __launch_bounds__(256) mgu_gemm_x(
    const unsigned short* __restrict__ e_bf,   // [S*B][E]
    const unsigned short* __restrict__ Wx_t,   // [2H][E] col-major weights
    const float* __restrict__ bz, const float* __restrict__ bh,
    unsigned short* __restrict__ xz,           // [S*B][H] bf16
    unsigned short* __restrict__ xh) {
  __shared__ unsigned smA[16 * 260];           // 16 x 512 halves, padded pitch
  const int tid = threadIdx.x;
  const int lane = tid & 31;
  const int wv = tid >> 5;
  const int m0 = blockIdx.y * 16;
  const int n0 = blockIdx.x * 128 + wv * 16;

  const unsigned* gA = (const unsigned*)(e_bf + (size_t)m0 * E_);
#pragma unroll 4
  for (int i = tid; i < 16 * (E_ / 2); i += 256) {
    int r = i >> 8, c = i & 255;
    smA[r * 260 + c] = gA[r * 256 + c];
  }
  __syncthreads();

  v8f acc = {};
  for (int k0 = 0; k0 < E_; k0 += 32) {
    // prefetch next K-chunk of this wave's weight column (global_prefetch_b8)
    if (k0 + 32 < E_) {
      const int col = n0 + (lane & 15);
      __builtin_prefetch(Wx_t + (size_t)col * E_ + k0 + 32, 0, 1);
    }
    v16bf a = frag_A_lds(smA, 260, k0, lane);
    v16bf b = frag_B_t(Wx_t, E_, k0, n0, lane);
    acc = WMMA_BF16(a, b, acc);
  }

  const int n = n0 + (lane & 15);
  const int mb = m0 + ((lane < 16) ? 0 : 8);
  const int nc = n & (H_ - 1);
  unsigned short* outp = (n < H_) ? xz : xh;
  const float bias = (n < H_) ? bz[nc] : bh[nc];
#pragma unroll
  for (int r = 0; r < 8; ++r)
    outp[(size_t)(mb + r) * H_ + nc] = f2bf(acc[r] + bias);
}

// =====================================================================
// Phase 2: one recurrence step.
//   pre_z = h@Wzh + xz[t] ; pre_h = h@Whh + xh[t]
//   h' = (1-sigmoid(pre_z))*h + sigmoid(pre_z)*tanh(pre_h)
// Block: 64 rows x 128 cols; 8 waves, each wave = 4 M-tiles x 1 N-tile x 2
// gates (8 WMMAs / K-chunk) so each B fragment is reused 4x.
// A tile staged into LDS with GLOBAL_LOAD_ASYNC_TO_LDS_B128 (ASYNCcnt).
// =====================================================================
#define STEP_PITCH_DW 516                       // 1024 halves + 4-dword pad
#define STEP_LDS_DW   (64 * STEP_PITCH_DW)

__global__ void __launch_bounds__(256) mgu_step(
    const unsigned short* __restrict__ h_bf_cur,  // [B][H] bf16
    const float* __restrict__ h_cur,              // [B][H] f32
    const unsigned short* __restrict__ Wr_t,      // [2H][H] col-major
    const unsigned short* __restrict__ xz,        // [S*B][H]
    const unsigned short* __restrict__ xh,
    float* __restrict__ h_nxt,
    unsigned short* __restrict__ h_bf_nxt,
    int t) {
  extern __shared__ unsigned smA[];               // 64 x 516 dwords
  const int tid = threadIdx.x;
  const int lane = tid & 31;
  const int wv = tid >> 5;
  const int m0 = blockIdx.y * 64;                 // grid.y = 2
  const int n0 = blockIdx.x * 128 + wv * 16;      // grid.x = 8

  // -- async stage A tile: 64 rows x 2048B, 16B chunks, padded LDS pitch --
  {
    const unsigned lds_base = lds_offset_of(smA);
    const char* gbase = (const char*)(h_bf_cur + (size_t)m0 * H_);
#pragma unroll 4
    for (int i = tid; i < 64 * 128; i += 256) {   // 8192 x b128
      int r = i >> 7, c = i & 127;
      unsigned ldsoff = lds_base + (unsigned)(r * (STEP_PITCH_DW * 4) + c * 16);
      unsigned long long g = (unsigned long long)(gbase + r * 2048 + c * 16);
      asm volatile("global_load_async_to_lds_b128 %0, %1, off"
                   :: "v"(ldsoff), "v"(g) : "memory");
    }
#if __has_builtin(__builtin_amdgcn_s_wait_asynccnt)
    __builtin_amdgcn_s_wait_asynccnt(0);
#else
    asm volatile("s_wait_asynccnt 0" ::: "memory");
#endif
    __syncthreads();
  }

  v8f accz[4] = {v8f{}, v8f{}, v8f{}, v8f{}};
  v8f acch[4] = {v8f{}, v8f{}, v8f{}, v8f{}};
  for (int k0 = 0; k0 < H_; k0 += 32) {
    v16bf bzf = frag_B_t(Wr_t, H_, k0, n0, lane);
    v16bf bhf = frag_B_t(Wr_t, H_, k0, n0 + H_, lane);
#pragma unroll
    for (int mt = 0; mt < 4; ++mt) {
      v16bf a = frag_A_lds(smA + mt * 16 * STEP_PITCH_DW, STEP_PITCH_DW, k0, lane);
      accz[mt] = WMMA_BF16(a, bzf, accz[mt]);
      acch[mt] = WMMA_BF16(a, bhf, acch[mt]);
    }
  }

  const int n = n0 + (lane & 15);
  const int rowoff = (lane < 16) ? 0 : 8;
  const size_t toff = (size_t)t * B_ * H_;
#pragma unroll
  for (int mt = 0; mt < 4; ++mt) {
#pragma unroll
    for (int r = 0; r < 8; ++r) {
      const int m = m0 + mt * 16 + rowoff + r;
      const size_t gi = (size_t)m * H_ + n;
      float zpre = accz[mt][r] + bf2f(xz[toff + gi]);
      float hpre = acch[mt][r] + bf2f(xh[toff + gi]);
      float z  = 1.0f / (1.0f + __expf(-zpre));
      float ht = tanhf(hpre);
      float hnew = (1.0f - z) * h_cur[gi] + z * ht;
      h_nxt[gi] = hnew;
      h_bf_nxt[gi] = f2bf(hnew);
    }
  }
}

// =====================================================================
// Phase 3: logits = h_final @ Wfc + bfc   (N padded to 1024, store-guarded)
// =====================================================================
__global__ void __launch_bounds__(256) mgu_fc(
    const unsigned short* __restrict__ h_bf,     // [B][H]
    const unsigned short* __restrict__ Wfc_t,    // [H pad][H] col-major
    const float* __restrict__ bfc,
    float* __restrict__ out) {                   // [B][NCLS]
  __shared__ unsigned smA[16 * 516];
  const int tid = threadIdx.x;
  const int lane = tid & 31;
  const int wv = tid >> 5;
  const int m0 = blockIdx.y * 16;
  const int n0 = blockIdx.x * 128 + wv * 16;

  const unsigned* gA = (const unsigned*)(h_bf + (size_t)m0 * H_);
#pragma unroll 4
  for (int i = tid; i < 16 * (H_ / 2); i += 256) {
    int r = i >> 9, c = i & 511;
    smA[r * 516 + c] = gA[r * 512 + c];
  }
  __syncthreads();

  v8f acc = {};
  for (int k0 = 0; k0 < H_; k0 += 32) {
    v16bf a = frag_A_lds(smA, 516, k0, lane);
    v16bf b = frag_B_t(Wfc_t, H_, k0, n0, lane);
    acc = WMMA_BF16(a, b, acc);
  }

  const int n = n0 + (lane & 15);
  const int mb = m0 + ((lane < 16) ? 0 : 8);
  if (n < NCLS_) {
    const float bias = bfc[n];
#pragma unroll
    for (int r = 0; r < 8; ++r)
      out[(size_t)(mb + r) * NCLS_ + n] = acc[r] + bias;
  }
}

// =====================================================================
extern "C" void kernel_launch(void* const* d_in, const int* in_sizes, int n_in,
                              void* d_out, int out_size, void* d_ws,
                              size_t ws_size, hipStream_t stream) {
  (void)in_sizes; (void)n_in; (void)out_size; (void)ws_size;
  const int*   x   = (const int*)d_in[0];
  const float* emb = (const float*)d_in[1];
  const float* Wz  = (const float*)d_in[2];
  const float* bz  = (const float*)d_in[3];
  const float* Wh  = (const float*)d_in[4];
  const float* bh  = (const float*)d_in[5];
  const float* Wfc = (const float*)d_in[6];
  const float* bfc = (const float*)d_in[7];
  float* out = (float*)d_out;

  char* p = (char*)d_ws;
  unsigned short* Wx_t  = (unsigned short*)p; p += (size_t)2 * H_ * E_ * 2;
  unsigned short* Wr_t  = (unsigned short*)p; p += (size_t)2 * H_ * H_ * 2;
  unsigned short* Wfc_t = (unsigned short*)p; p += (size_t)H_ * H_ * 2;
  unsigned short* e_bf  = (unsigned short*)p; p += (size_t)S_ * B_ * E_ * 2;
  unsigned short* xz    = (unsigned short*)p; p += (size_t)S_ * B_ * H_ * 2;
  unsigned short* xh    = (unsigned short*)p; p += (size_t)S_ * B_ * H_ * 2;
  float*          h32   = (float*)p;          p += (size_t)2 * B_ * H_ * 4;
  unsigned short* hbf   = (unsigned short*)p; p += (size_t)2 * B_ * H_ * 2;

  // weights -> bf16 transposed
  mgu_prep_w<<<(2 * H_ * E_ + 2 * H_ * H_ + H_ * H_) / 256, 256, 0, stream>>>(
      Wz, Wh, Wfc, Wx_t, Wr_t, Wfc_t);
  // embedding gather
  mgu_embed<<<S_ * B_, 256, 0, stream>>>(x, emb, e_bf);
  // h0 = 0
  mgu_init_h<<<(B_ * H_ + 255) / 256, 256, 0, stream>>>(h32, hbf);
  // big input-projection GEMM (parallel part of the recurrence)
  mgu_gemm_x<<<dim3(16, (S_ * B_) / 16), 256, 0, stream>>>(e_bf, Wx_t, bz, bh,
                                                           xz, xh);
  // sequential recurrence, ping-pong h buffers
  const size_t stepLds = (size_t)STEP_LDS_DW * 4;   // 132 KB (of 320 KB/WGP)
  for (int t = 0; t < S_; ++t) {
    const size_t cur = (size_t)(t & 1) * B_ * H_;
    const size_t nxt = (size_t)((t + 1) & 1) * B_ * H_;
    mgu_step<<<dim3(8, 2), 256, stepLds, stream>>>(hbf + cur, h32 + cur, Wr_t,
                                                   xz, xh, h32 + nxt, hbf + nxt,
                                                   t);
  }
  // classifier (final h lives in buffer 0 after 512 steps)
  mgu_fc<<<dim3(8, 8), 256, 0, stream>>>(hbf, Wfc_t, bfc, out);
}
